// PCRefinement_15668040696146
// MI455X (gfx1250) — matrix operand order repaired and checked by reading, not compiled, not run on hardware
//
#include <hip/hip_runtime.h>

// ---------------------------------------------------------------------------
// PCRefinement for MI455X (gfx1250, wave32).
//  Phase 1: 1-NN as argmax(q.v - 0.5|v|^2) via V_WMMA_F32_16X16X4_F32 (K=3->4),
//           voxel stream staged through LDS with async global->LDS copies
//           (ASYNCcnt-tracked, double buffered, overlapped with compute).
//  Phase 2: dense scatter-max pooling with integer-trick float atomic max.
//  Phase 3: voxel MLP as bf16 V_WMMA_F32_16X16X32_BF16 GEMM (16x64 strip per
//           wave for 4x A-reuse), fused bias+ReLU+occ-mask+transposed store.
// ---------------------------------------------------------------------------

typedef float   v2f   __attribute__((ext_vector_type(2)));
typedef float   v8f   __attribute__((ext_vector_type(8)));
typedef __bf16  bf16_t;
typedef bf16_t  v16bf __attribute__((ext_vector_type(16)));
typedef int     gvi4  __attribute__((__vector_size__(16)));  // matches builtin param

#define PI_F 3.14159265358979f
static constexpr int NPTS = 120000;
#define KNN_CHUNK 64            // voxel tiles per LDS chunk: 64*256B = 16 KB

__device__ __forceinline__ unsigned int f32_to_bf16(float f) {
  unsigned int u = __float_as_uint(f);
  u += 0x7fffu + ((u >> 16) & 1u);            // round-to-nearest-even
  return u >> 16;
}

// IEEE float max via ordered-integer atomics (maps to global_atomic_max/min).
__device__ __forceinline__ void atomic_max_f32(float* addr, float v) {
  if (v >= 0.0f) atomicMax((int*)addr, __float_as_int(v));
  else           atomicMin((unsigned int*)addr, (unsigned int)__float_as_int(v));
}

// 16B global -> LDS copy on the CDNA5 async path when available.
__device__ __forceinline__ void async_cp16(const uint4* g, uint4* l) {
#if __has_builtin(__builtin_amdgcn_global_load_async_to_lds_b128)
  __builtin_amdgcn_global_load_async_to_lds_b128((gvi4*)g, (gvi4*)l, 0, 0);
#else
  *l = *g;                                    // fallback: load + ds_store
#endif
}

template <int N>
__device__ __forceinline__ void wait_async() {
#if __has_builtin(__builtin_amdgcn_s_wait_asynccnt)
  __builtin_amdgcn_s_wait_asynccnt(N);
#else
  if constexpr (N == 0) asm volatile("s_wait_asynccnt 0x0" ::: "memory");
  else                  asm volatile("s_wait_asynccnt 0x4" ::: "memory");
#endif
}

// ---------------------------------------------------------------------------
// Fill vmax with -1e30 (identity for max) and occ with 0.
__global__ void pcr_fill(float* vmax, int* occ, long long nV, int nOcc) {
  long long i = (long long)blockIdx.x * blockDim.x + threadIdx.x;
  if (i < nV) vmax[i] = -1.0e30f;
  if (i < nOcc) occ[i] = 0;
}

// ---------------------------------------------------------------------------
// Pack voxel coords into the f32 16x4 A-fragment layout:
//  lane r (r<16): (x, y) = K0,K1 ; lane 16+r: (z, -|v|^2/2) = K2,K3
// Rows j>=M (chunk padding included) score -1e30 so they never win argmax.
__global__ void pcr_prep_knnA(const int* __restrict__ vidx,
                              float* __restrict__ packedA, int M, int MPC) {
  int j = blockIdx.x * blockDim.x + threadIdx.x;
  if (j >= MPC) return;
  float x = 0.f, y = 0.f, z = 0.f, w = -1.0e30f;
  if (j < M) {
    x = (float)vidx[j * 4 + 1];
    y = (float)vidx[j * 4 + 2];
    z = (float)vidx[j * 4 + 3];
    w = -0.5f * (x * x + y * y + z * z);
  }
  int t = j >> 4, r = j & 15;
  float2* A = (float2*)packedA;
  A[t * 32 + r]      = make_float2(x, y);
  A[t * 32 + 16 + r] = make_float2(z, w);
}

// ---------------------------------------------------------------------------
// Copy one 16 KB chunk (KNN_CHUNK voxel tiles) to LDS: 4 async b128 per thread.
__device__ __forceinline__ void knn_copy_chunk(const uint4* __restrict__ gsrc,
                                               uint4* lbuf, int tid) {
#pragma unroll
  for (int i = 0; i < 4; ++i)
    async_cp16(gsrc + tid + i * 256, lbuf + tid + i * 256);
}

// ---------------------------------------------------------------------------
// 1-NN: one wave per 16-query tile; 8 waves/block share the LDS-staged voxel
// stream. Inner loop: ds_load_b64 -> wmma_f32_16x16x4_f32 -> 8 max/select.
__global__ void pcr_knn_wmma(const float* __restrict__ pcoord,
                             const uint4* __restrict__ packedA,
                             int* __restrict__ p2p,
                             int nQTiles, int nChunks, int N,
                             float pvs0, float pvs1, float pvs2) {
  __shared__ uint4 abuf[2][KNN_CHUNK * 16];
  int tid = threadIdx.x;
  int lane = tid & 31;
  int wave = blockIdx.x * (blockDim.x >> 5) + (tid >> 5);
  bool active = wave < nQTiles;

  v2f bfrag; bfrag.x = 0.0f; bfrag.y = 0.0f;
  int qIdx = 0;
  if (active) {
    qIdx = wave * 16 + (lane & 15);
    int qc = qIdx < N ? qIdx : N - 1;
    float qx = floorf((pcoord[qc * 3 + 0] - 0.0f) / pvs0);
    float qy = floorf((pcoord[qc * 3 + 1] + PI_F) / pvs1);
    float qz = floorf((pcoord[qc * 3 + 2] + 4.0f) / pvs2);
    if (lane < 16) { bfrag.x = qx; bfrag.y = qy; }   // K0,K1
    else           { bfrag.x = qz; bfrag.y = 1.0f; } // K2,K3 (bias column)
  }

  float best = -3.0e38f;
  int bestIdx = 0;
  const float2* lds2 = (const float2*)abuf;
  int half8 = (lane >> 4) << 3;

  knn_copy_chunk(packedA, &abuf[0][0], tid);
  for (int c = 0; c < nChunks; ++c) {
    if (c + 1 < nChunks) {
      knn_copy_chunk(packedA + (size_t)(c + 1) * (KNN_CHUNK * 16),
                     &abuf[(c + 1) & 1][0], tid);
      wait_async<4>();          // chunk c resident; chunk c+1 still in flight
    } else {
      wait_async<0>();
    }
    __syncthreads();
    if (active) {
      const float2* src = lds2 + (size_t)(c & 1) * (KNN_CHUNK * 32);
      int tbase = c * (KNN_CHUNK * 16) + half8;
      for (int t = 0; t < KNN_CHUNK; ++t) {
        float2 a2 = src[t * 32 + lane];
        v2f afrag; afrag.x = a2.x; afrag.y = a2.y;
        v8f d = {};
        d = __builtin_amdgcn_wmma_f32_16x16x4_f32(
            false, afrag, false, bfrag, (short)0, d, false, false);
        int base = tbase + t * 16;
#pragma unroll
        for (int r = 0; r < 8; ++r) {
          float sc = d[r];
          if (sc > best) { best = sc; bestIdx = base + r; }
        }
      }
    }
    __syncthreads();
  }

  if (active) {
    float ob = __shfl_xor(best, 16, 32);
    int   oi = __shfl_xor(bestIdx, 16, 32);
    if (ob > best) { best = ob; bestIdx = oi; }
    if (lane < 16 && qIdx < N) p2p[qIdx] = bestIdx;
  }
}

// ---------------------------------------------------------------------------
// Per-point cartesian voxel linearization + occupancy flag.
__global__ void pcr_prep_lin(const float* __restrict__ coord,
                             int* __restrict__ lin, int* __restrict__ occ,
                             int N, int D0, int D1, int D2,
                             float s0, float s1, float s2) {
  int n = blockIdx.x * blockDim.x + threadIdx.x;
  if (n >= N) return;
  int q0 = (int)floorf((coord[n * 3 + 0] + 25.6f) * s0);
  int q1 = (int)floorf((coord[n * 3 + 1] + 25.6f) * s1);
  int q2 = (int)floorf((coord[n * 3 + 2] + 2.0f)  * s2);
  q0 = q0 < 0 ? 0 : (q0 > D0 - 1 ? D0 - 1 : q0);
  q1 = q1 < 0 ? 0 : (q1 > D1 - 1 ? D1 - 1 : q1);
  q2 = q2 < 0 ? 0 : (q2 > D2 - 1 ? D2 - 1 : q2);
  int l = (q0 * D1 + q1) * D2 + q2;
  lin[n] = l;
  occ[l] = 1;
}

// ---------------------------------------------------------------------------
// Scatter-max of concatenated [point_feat | vfeat[p2p]] into dense G x K.
__global__ void pcr_scatter_max(const float* __restrict__ pfeat,
                                const float* __restrict__ vfeat,
                                const int* __restrict__ p2p,
                                const int* __restrict__ lin,
                                float* __restrict__ vmax,
                                int N, int K, int C) {
  long long gid = (long long)blockIdx.x * blockDim.x + threadIdx.x;
  if (gid >= (long long)N * K) return;
  int n = (int)(gid / K);
  int k = (int)(gid - (long long)n * K);
  float v = (k < 256) ? pfeat[(long long)n * 256 + k]
                      : vfeat[(long long)p2p[n] * C + (k - 256)];
  atomic_max_f32(vmax + (long long)lin[n] * K + k, v);
}

// ---------------------------------------------------------------------------
// Pre-pack W (f32, K x C row-major) into the bf16 B-fragment layout.
__global__ void pcr_prep_W(const float* __restrict__ W,
                           unsigned int* __restrict__ pW,
                           int K, int C, int tilesN) {
  int e = blockIdx.x * blockDim.x + threadIdx.x;
  int total = (K >> 5) * tilesN * 256;
  if (e >= total) return;
  int v    = e & 7;
  int lane = (e >> 3) & 31;
  int rest = e >> 8;
  int tn = rest % tilesN;
  int kb = rest / tilesN;
  int col = tn * 16 + (lane & 15);
  int k0  = kb * 32 + ((lane >> 4) << 3) + ((v < 4) ? 2 * v : 16 + 2 * (v - 4));
  unsigned int lo = f32_to_bf16(W[(long long)k0 * C + col]);
  unsigned int hi = f32_to_bf16(W[(long long)(k0 + 1) * C + col]);
  pW[e] = lo | (hi << 16);
}

// ---------------------------------------------------------------------------
// Dense GEMM: one wave computes a 16x64 output strip (4 N-tiles) so each A
// fragment (f32 -> bf16 on the fly) feeds 4 wmma_f32_16x16x32_bf16 ops.
__global__ void pcr_gemm_wmma(const float* __restrict__ vmax,
                              const unsigned int* __restrict__ pW,
                              const float* __restrict__ bias,
                              const int* __restrict__ occ,
                              float* __restrict__ out,
                              int tilesM, int tnGroups, int tilesN,
                              int K, int C, int D0, int D1, int D2) {
  int wave = blockIdx.x * (blockDim.x >> 5) + (threadIdx.x >> 5);
  int lane = threadIdx.x & 31;
  if (wave >= tilesM * tnGroups) return;
  int tm = wave / tnGroups;
  int tn0 = (wave - tm * tnGroups) * 4;

  int row = tm * 16 + (lane & 15);
  int kHalf = (lane >> 4) << 3;           // A K-offset: 0 or 8
  const float* arow = vmax + (long long)row * K;
  int nkb = K >> 5;

  v8f acc[4] = {};
  for (int kb = 0; kb < nkb; ++kb) {
    int kbase = kb * 32 + kHalf;
    __builtin_prefetch(arow + kbase + 64, 0, 1);
    union { v16bf vec; unsigned int u[8]; } af;
#pragma unroll
    for (int vi = 0; vi < 8; ++vi) {
      int k0 = kbase + ((vi < 4) ? 2 * vi : 16 + 2 * (vi - 4));
      float2 f = *(const float2*)(arow + k0);
      af.u[vi] = f32_to_bf16(f.x) | (f32_to_bf16(f.y) << 16);
    }
    // per-(kb,tn) fragment block: 256 uints = 64 uint4; per-lane 2 uint4
    const uint4* wrow =
        (const uint4*)pW + (((long long)kb * tilesN + tn0) * 32 + lane) * 2;
#pragma unroll
    for (int j = 0; j < 4; ++j) {
      union { v16bf vec; uint4 q[2]; } bfr;
      bfr.q[0] = wrow[j * 64 + 0];
      bfr.q[1] = wrow[j * 64 + 1];
      acc[j] = __builtin_amdgcn_wmma_f32_16x16x32_bf16(
          false, af.vec, false, bfr.vec, (short)0, acc[j], false, false);
    }
  }

  int mOff = (lane >> 4) << 3;
#pragma unroll
  for (int j = 0; j < 4; ++j) {
    int col = (tn0 + j) * 16 + (lane & 15);
    float bc = bias[col];
#pragma unroll
    for (int r = 0; r < 8; ++r) {
      int g = tm * 16 + mOff + r;
      float h = 0.0f;
      if (occ[g]) {
        h = acc[j][r] + bc;
        h = h < 0.0f ? 0.0f : h;
      }
      int d2 = g % D2;
      int t  = g / D2;
      int d1 = t % D1;
      int d0 = t / D1;
      out[(((long long)col * D2 + d2) * D0 + d0) * D1 + d1] = h;
    }
  }
}

// ---------------------------------------------------------------------------
static inline size_t align256(size_t x) { return (x + 255) & ~(size_t)255; }

static void run_level(int level,
                      const float* pfeat, const float* pcoord,
                      const float* coord, const int* vidx, const float* vfeat,
                      const float* W, const float* bias,
                      float* out, void* ws, hipStream_t stream) {
  const int Ms[3] = {40000, 15000, 5000};
  int s  = 2 << level;                 // 2^(level+1)
  int C  = 128 << level;
  int K  = 256 + C;
  int M  = Ms[level];
  int MP = (M + 15) & ~15;
  int tilesA  = MP / 16;
  int nChunks = (tilesA + KNN_CHUNK - 1) / KNN_CHUNK;
  int MPC     = nChunks * KNN_CHUNK * 16;   // chunk-padded voxel rows
  int D0 = 256 / s, D1 = 256 / s, D2 = 32 / s;
  int G  = D0 * D1 * D2;
  int tilesM = G / 16, tilesN = C / 16, tnGroups = tilesN / 4;
  int N = NPTS;

  float pvs0 = (50.0f - 0.0f)   / ((480.0f - 1.0f) / (float)s);
  float pvs1 = (PI_F - (-PI_F)) / ((360.0f - 1.0f) / (float)s);
  float pvs2 = (2.0f - (-4.0f)) / ((32.0f  - 1.0f) / (float)s);
  float cs0 = (float)D0 / 51.2f;
  float cs1 = (float)D1 / 51.2f;
  float cs2 = (float)D2 / 6.4f;

  // workspace carve (regions reused across levels; peak ~410 MB at level 0)
  char* p = (char*)ws;
  int*   p2p     = (int*)p;            p += align256((size_t)N * 4);
  int*   lin     = (int*)p;            p += align256((size_t)N * 4);
  float* packedA = (float*)p;          p += align256((size_t)MPC * 4 * 4);
  unsigned int* pW = (unsigned int*)p; p += align256((size_t)K * C / 2 * 4);
  int*   occ     = (int*)p;            p += align256((size_t)G * 4);
  float* vmax    = (float*)p;

  long long nV = (long long)G * K;
  pcr_fill<<<(int)((nV + 255) / 256), 256, 0, stream>>>(vmax, occ, nV, G);
  pcr_prep_knnA<<<(MPC + 255) / 256, 256, 0, stream>>>(vidx, packedA, M, MPC);
  {
    int total = (K >> 5) * tilesN * 256;
    pcr_prep_W<<<(total + 255) / 256, 256, 0, stream>>>(W, pW, K, C, tilesN);
  }
  {
    int nQTiles = (N + 15) / 16;
    int blocks = (nQTiles + 7) / 8;     // 8 waves / block
    pcr_knn_wmma<<<blocks, 256, 0, stream>>>(pcoord, (const uint4*)packedA,
                                             p2p, nQTiles, nChunks, N,
                                             pvs0, pvs1, pvs2);
  }
  pcr_prep_lin<<<(N + 255) / 256, 256, 0, stream>>>(coord, lin, occ, N,
                                                    D0, D1, D2, cs0, cs1, cs2);
  {
    long long thr = (long long)N * K;
    pcr_scatter_max<<<(int)((thr + 255) / 256), 256, 0, stream>>>(
        pfeat, vfeat, p2p, lin, vmax, N, K, C);
  }
  {
    int waves = tilesM * tnGroups;
    int blocks = (waves + 7) / 8;       // 8 waves / block
    pcr_gemm_wmma<<<blocks, 256, 0, stream>>>(vmax, pW, bias, occ, out,
                                              tilesM, tnGroups, tilesN,
                                              K, C, D0, D1, D2);
  }
}

extern "C" void kernel_launch(void* const* d_in, const int* in_sizes, int n_in,
                              void* d_out, int out_size, void* d_ws,
                              size_t ws_size, hipStream_t stream) {
  // setup_inputs() dict order:
  // 0 point_feat, 1 p_coord, 2 coord, 3 batch,
  // 4 idx0, 5 f0, 6 W0, 7 b0, 8 idx1, 9 f1, 10 W1, 11 b1,
  // 12 idx2, 13 f2, 14 W2, 15 b2
  const float* pfeat  = (const float*)d_in[0];
  const float* pcoord = (const float*)d_in[1];
  const float* coord  = (const float*)d_in[2];
  const int*   idx[3]  = {(const int*)d_in[4],  (const int*)d_in[8],  (const int*)d_in[12]};
  const float* fv[3]   = {(const float*)d_in[5],(const float*)d_in[9],(const float*)d_in[13]};
  const float* Wl[3]   = {(const float*)d_in[6],(const float*)d_in[10],(const float*)d_in[14]};
  const float* bl[3]   = {(const float*)d_in[7],(const float*)d_in[11],(const float*)d_in[15]};

  float* out = (float*)d_out;
  size_t off = 0;
  for (int L = 0; L < 3; ++L) {
    int s = 2 << L;
    int C = 128 << L;
    int G = (256 / s) * (256 / s) * (32 / s);
    run_level(L, pfeat, pcoord, coord, idx[L], fv[L], Wl[L], bl[L],
              out + off, d_ws, stream);
    off += (size_t)G * C;
  }
}
